// g_cheb_gcn_54709293416499
// MI455X (gfx1250) — matrix operand order repaired and checked by reading, not compiled, not run on hardware
//
#include <hip/hip_runtime.h>
#include <hip/hip_bf16.h>
#include <math.h>

// Problem constants (match reference)
#define BB 8
#define NN 2000
#define NP 2048            // NN padded to multiple of 64 (K-dim AND col-dim padding, zero filled)
#define CC 64
#define HH 64
#define DD 16
#define KC 2               // cheb_k
#define BN (BB*NN)         // 16000
#define OC (CC*CC)         // 4096

typedef __attribute__((ext_vector_type(16))) _Float16 v16h;
typedef __attribute__((ext_vector_type(8)))  _Float16 v8h;
typedef __attribute__((ext_vector_type(8)))  float    v8f;

__device__ __forceinline__ v8f zero8() {
  v8f r;
#pragma unroll
  for (int i = 0; i < 8; ++i) r[i] = 0.f;
  return r;
}

// ---- WMMA fragment loaders (CDNA5 wave32 layouts, CDNA5 ISA §7.12.2) ----
// A (16x32 f16): lane m=lane&15, half=lane>>4; elements 0..7 hold k = k0+8*half+e,
// elements 8..15 hold k = k0+16+8*half+e  -> two contiguous 16B loads.
__device__ __forceinline__ v16h load_a_frag(const _Float16* A, int lda) {
  const int lane = threadIdx.x & 31;
  const _Float16* p = A + (size_t)(lane & 15) * lda + ((lane >> 4) << 3);
  v8h lo = *(const v8h*)p;
  v8h hi = *(const v8h*)(p + 16);
  return __builtin_shufflevector(lo, hi, 0,1,2,3,4,5,6,7,8,9,10,11,12,13,14,15);
}
// B (32x16 f16) from K-major storage Bt[col][k]: lane col=lane&15, half=lane>>4,
// elements 0..7 hold k = k0+8*half+e, elements 8..15 hold k = k0+16+8*half+e -> two 16B loads.
__device__ __forceinline__ v16h load_b_frag(const _Float16* Bt, int ldb) {
  const int lane = threadIdx.x & 31;
  const _Float16* p = Bt + (size_t)(lane & 15) * ldb + ((lane >> 4) << 3);
  v8h lo = *(const v8h*)p;
  v8h hi = *(const v8h*)(p + 16);
  return __builtin_shufflevector(lo, hi, 0,1,2,3,4,5,6,7,8,9,10,11,12,13,14,15);
}

template<int EPI>
__device__ __forceinline__ void store_tile(float* __restrict__ out, long long out_rs,
                                           int mrow, int nc, const float* __restrict__ bias,
                                           v8f acc) {
  const float bv = (EPI != 0) ? bias[nc] : 0.f;
#pragma unroll
  for (int e = 0; e < 8; ++e) {
    float v = acc[e];
    if (EPI == 1) v = fmaxf(v + bv, 0.f);
    if (EPI == 2) v = 0.1f * tanhf(v + bv);
    out[(size_t)(mrow + e) * out_rs + nc] = v;
  }
}

// ---- Generic WMMA GEMM: out[M x (gridDim.y*64)] = A[M x K] * Bt[cols x K]^T ----
// EPI: 0 = plain f32 store, 1 = bias+relu, 2 = 0.1*tanh(x+bias)
// One wave computes a 16x64 tile. Two-stage software pipeline over K (Kdim % 64 == 0):
// explicit two-set register rotation -> no fragment copies, loads overlap WMMA.
template<int EPI>
__global__ __launch_bounds__(32)
void k_gemm(const _Float16* __restrict__ A, int lda, long long a_bs,
            const _Float16* __restrict__ Bt, int ldb, long long b_bs,
            const float* __restrict__ bias,
            float* __restrict__ out, long long out_rs, long long out_bs,
            int Kdim)
{
  const int lane = threadIdx.x & 31;
  const int m0 = blockIdx.x << 4;
  const int n0 = blockIdx.y << 6;
  const int bz = blockIdx.z;
  A   += (size_t)bz * a_bs + (size_t)m0 * lda;
  const _Float16* B0 = Bt + (size_t)bz * b_bs + (size_t)(n0     ) * ldb;
  const _Float16* B1 = Bt + (size_t)bz * b_bs + (size_t)(n0 + 16) * ldb;
  const _Float16* B2 = Bt + (size_t)bz * b_bs + (size_t)(n0 + 32) * ldb;
  const _Float16* B3 = Bt + (size_t)bz * b_bs + (size_t)(n0 + 48) * ldb;
  out += (size_t)bz * out_bs;

  v8f acc0 = zero8(), acc1 = zero8(), acc2 = zero8(), acc3 = zero8();

#define LOAD_SET(va, vb0, vb1, vb2, vb3, off)        \
  va  = load_a_frag(A  + (off), lda);                \
  vb0 = load_b_frag(B0 + (off), ldb);                \
  vb1 = load_b_frag(B1 + (off), ldb);                \
  vb2 = load_b_frag(B2 + (off), ldb);                \
  vb3 = load_b_frag(B3 + (off), ldb)

#define WMMA_SET(va, vb0, vb1, vb2, vb3)                                                              \
  acc0 = __builtin_amdgcn_wmma_f32_16x16x32_f16(false, va, false, vb0, (short)0, acc0, false, false); \
  acc1 = __builtin_amdgcn_wmma_f32_16x16x32_f16(false, va, false, vb1, (short)0, acc1, false, false); \
  acc2 = __builtin_amdgcn_wmma_f32_16x16x32_f16(false, va, false, vb2, (short)0, acc2, false, false); \
  acc3 = __builtin_amdgcn_wmma_f32_16x16x32_f16(false, va, false, vb3, (short)0, acc3, false, false)

  v16h a0, b00, b01, b02, b03;
  v16h a1, b10, b11, b12, b13;
  LOAD_SET(a0, b00, b01, b02, b03, 0);
  LOAD_SET(a1, b10, b11, b12, b13, 32);

  int k0 = 0;
  for (; k0 + 128 <= Kdim; k0 += 64) {
    WMMA_SET(a0, b00, b01, b02, b03);            // consume set0 (k0)
    LOAD_SET(a0, b00, b01, b02, b03, k0 + 64);   // refill set0 for k0+64 (overlaps next WMMAs)
    __builtin_prefetch(A + k0 + 128, 0, 3);      // near-scope global_prefetch_b8
    WMMA_SET(a1, b10, b11, b12, b13);            // consume set1 (k0+32)
    LOAD_SET(a1, b10, b11, b12, b13, k0 + 96);   // refill set1 for k0+96
  }
  WMMA_SET(a0, b00, b01, b02, b03);              // Kdim-64
  WMMA_SET(a1, b10, b11, b12, b13);              // Kdim-32

#undef LOAD_SET
#undef WMMA_SET

  // C/D layout: VGPR e holds row m0 + 8*(lane>>4) + e, col = lane&15
  const int ncol = lane & 15;
  const int mrow = m0 + ((lane >> 4) << 3);
  store_tile<EPI>(out, out_rs, mrow, n0 +  0 + ncol, bias, acc0);
  store_tile<EPI>(out, out_rs, mrow, n0 + 16 + ncol, bias, acc1);
  store_tile<EPI>(out, out_rs, mrow, n0 + 32 + ncol, bias, acc2);
  store_tile<EPI>(out, out_rs, mrow, n0 + 48 + ncol, bias, acc3);
}

// ---- f32 -> f16 copy with zero column padding ----
__global__ void k_cvt_pad(const float* __restrict__ src, _Float16* __restrict__ dst,
                          int rows, int cols, int ldd) {
  int i = blockIdx.x * 256 + threadIdx.x;
  if (i >= rows * ldd) return;
  int r = i / ldd, c = i % ldd;
  dst[i] = (c < cols) ? (_Float16)src[r * cols + c] : (_Float16)0.f;
}

__device__ __forceinline__ float blk_reduce(float v, float* red, bool do_max) {
  int t = threadIdx.x;
  red[t] = v; __syncthreads();
  for (int s = 128; s > 0; s >>= 1) {
    if (t < s) red[t] = do_max ? fmaxf(red[t], red[t + s]) : (red[t] + red[t + s]);
    __syncthreads();
  }
  float r = red[0]; __syncthreads();
  return r;
}

// ---- A = softmax(relu(emb @ emb^T), axis=1) -> f16, cols padded to NP with 0 ----
__global__ __launch_bounds__(256) void k_adj(const float* __restrict__ emb,
                                             _Float16* __restrict__ Aout) {
  __shared__ float row[NN];
  __shared__ float red[256];
  __shared__ float erow[DD];
  const int n = blockIdx.x, t = threadIdx.x;
  if (t < DD) erow[t] = emb[n * DD + t];
  __syncthreads();
  float lmax = 0.f;  // relu output >= 0
  for (int m = t; m < NN; m += 256) {
    float v = 0.f;
#pragma unroll
    for (int d = 0; d < DD; ++d) v += erow[d] * emb[m * DD + d];
    v = fmaxf(v, 0.f);
    row[m] = v;
    lmax = fmaxf(lmax, v);
  }
  const float mx = blk_reduce(lmax, red, true);
  float lsum = 0.f;
  for (int m = t; m < NN; m += 256) {
    float e = __expf(row[m] - mx);
    row[m] = e;
    lsum += e;
  }
  const float inv = 1.f / blk_reduce(lsum, red, false);
  for (int m = t; m < NP; m += 256)
    Aout[(size_t)n * NP + m] = (m < NN) ? (_Float16)(row[m] * inv) : (_Float16)0.f;
}

// ---- row softmax of T (2000 x NP f32, valid cols < NN) -> E f16 (ld NP, pad 0) ----
__global__ __launch_bounds__(256) void k_softmax_T(const float* __restrict__ T,
                                                   _Float16* __restrict__ E) {
  __shared__ float row[NN];
  __shared__ float red[256];
  const int n = blockIdx.x, t = threadIdx.x;
  const float* Tr = T + (size_t)n * NP;
  float lmax = -1e30f;
  for (int p = t; p < NN; p += 256) { float v = Tr[p]; row[p] = v; lmax = fmaxf(lmax, v); }
  const float mx = blk_reduce(lmax, red, true);
  float lsum = 0.f;
  for (int p = t; p < NN; p += 256) { float e = __expf(row[p] - mx); row[p] = e; lsum += e; }
  const float inv = 1.f / blk_reduce(lsum, red, false);
  for (int p = t; p < NP; p += 256)
    E[(size_t)n * NP + p] = (p < NN) ? (_Float16)(row[p] * inv) : (_Float16)0.f;
}

// ---- e1 = h.w1, e2 = h.w2 per (b,n) ----
__global__ void k_e12(const float* __restrict__ h, const float* __restrict__ w1,
                      const float* __restrict__ w2, float* __restrict__ e1,
                      float* __restrict__ e2) {
  int i = blockIdx.x * 256 + threadIdx.x;
  if (i >= BN) return;
  const float* hr = h + (size_t)i * HH;
  float a = 0.f, b = 0.f;
#pragma unroll
  for (int c = 0; c < HH; ++c) { a += hr[c] * w1[c]; b += hr[c] * w2[c]; }
  e1[i] = a; e2[i] = b;
}

// ---- (B, NN, HH) f32 -> K-major f16 (B, HH, NP), rows >= NN zero ----
__global__ void k_colmajor(const float* __restrict__ src, _Float16* __restrict__ dst) {
  int b = blockIdx.y;
  int i = blockIdx.x * 256 + threadIdx.x;
  if (i >= HH * NP) return;
  int c = i / NP, n = i % NP;
  float v = (n < NN) ? src[((size_t)b * NN + n) * HH + c] : 0.f;
  dst[(size_t)b * HH * NP + i] = (_Float16)v;
}

// ---- sigT[b][p][m] = sigmoid(e1[b,m]*e2[b,p] + bs[m,p]); pads -> 0 ----
// Slab per batch: NP rows (p) x NP cols (m), fully initialized.
__global__ void k_sig(const float* __restrict__ e1, const float* __restrict__ e2,
                      const float* __restrict__ bs, _Float16* __restrict__ sigT) {
  int b = blockIdx.y;
  int i = blockIdx.x * 256 + threadIdx.x;
  if (i >= NP * NP) return;
  int p = i / NP, m = i % NP;
  float v = 0.f;
  if (p < NN && m < NN) {
    float x = e1[b * NN + m] * e2[b * NN + p] + bs[(size_t)m * NN + p];
    v = 1.f / (1.f + __expf(-x));
  }
  sigT[(size_t)b * NP * NP + i] = (_Float16)v;
}

// ---- node-adaptive weights: w[n, j] = sum_d emb[n,d]*wp[d, j], j in [0, KC*HH*HH) ----
__global__ void k_weights(const float* __restrict__ emb, const float* __restrict__ wp,
                          _Float16* __restrict__ w) {
  int i = blockIdx.x * 256 + threadIdx.x;
  const int J = KC * HH * HH;  // 8192
  if (i >= NN * J) return;
  int n = i / J, j = i % J;
  float a = 0.f;
#pragma unroll
  for (int d = 0; d < DD; ++d) a += emb[n * DD + d] * wp[(size_t)d * J + j];
  w[(size_t)i] = (_Float16)a;
}

// ---- bias_node[n,o] = sum_d emb[n,d]*bp[d,o] ----
__global__ void k_biasnode(const float* __restrict__ emb, const float* __restrict__ bp,
                           float* __restrict__ bn) {
  int i = blockIdx.x * 256 + threadIdx.x;
  if (i >= NN * HH) return;
  int n = i / HH, o = i % HH;
  float a = 0.f;
#pragma unroll
  for (int d = 0; d < DD; ++d) a += emb[n * DD + d] * bp[d * HH + o];
  bn[i] = a;
}

// ---- g[bn,o] = bias_node[n,o] + sum_ki xg[bn,ki]*w[n,ki,o]; also f16 copy ----
__global__ __launch_bounds__(64) void k_g(const float* __restrict__ xg,
                                          const _Float16* __restrict__ w,
                                          const float* __restrict__ bnv,
                                          float* __restrict__ g,
                                          _Float16* __restrict__ g16) {
  __shared__ float xr[KC * HH];  // 128
  const int bn_i = blockIdx.x;   // b*NN + n
  const int o = threadIdx.x;     // 0..63
  const float* xrow = xg + (size_t)bn_i * (KC * HH);
  xr[o] = xrow[o];
  xr[o + HH] = xrow[o + HH];
  __syncthreads();
  const int n = bn_i % NN;
  const _Float16* wn = w + (size_t)n * (KC * HH * HH);
  float acc = bnv[n * HH + o];
#pragma unroll 8
  for (int ki = 0; ki < KC * HH; ++ki) acc += xr[ki] * (float)wn[ki * HH + o];
  g[(size_t)bn_i * HH + o] = acc;
  g16[(size_t)bn_i * HH + o] = (_Float16)acc;
}

static inline size_t alignup(size_t x) { return (x + 255) & ~(size_t)255; }

extern "C" void kernel_launch(void* const* d_in, const int* in_sizes, int n_in,
                              void* d_out, int out_size, void* d_ws, size_t ws_size,
                              hipStream_t stream) {
  const float* z     = (const float*)d_in[0];
  const float* W_in  = (const float*)d_in[1];
  const float* b_in  = (const float*)d_in[2];
  const float* W_out = (const float*)d_in[3];
  const float* b_out = (const float*)d_in[4];
  const float* emb   = (const float*)d_in[5];
  const float* wp    = (const float*)d_in[6];
  const float* bp    = (const float*)d_in[7];
  const float* w1    = (const float*)d_in[8];
  const float* w2    = (const float*)d_in[9];
  const float* vs    = (const float*)d_in[10];
  const float* bs    = (const float*)d_in[11];
  float* out = (float*)d_out;

  // workspace carve-up
  char* w = (char*)d_ws;
  auto alloc = [&](size_t bytes) { char* p = w; w += alignup(bytes); return p; };
  _Float16* z16    = (_Float16*)alloc((size_t)BN * CC * 2);
  _Float16* Win16  = (_Float16*)alloc((size_t)HH * CC * 2);
  _Float16* Wout16 = (_Float16*)alloc((size_t)OC * HH * 2);
  _Float16* vs16   = (_Float16*)alloc((size_t)NN * NP * 2);
  _Float16* A16    = (_Float16*)alloc((size_t)NN * NP * 2);
  float*    h      = (float*)   alloc((size_t)BN * HH * 4);
  _Float16* hT     = (_Float16*)alloc((size_t)BB * HH * NP * 2);
  float*    e1     = (float*)   alloc((size_t)BN * 4);
  float*    e2     = (float*)   alloc((size_t)BN * 4);
  _Float16* sigT   = (_Float16*)alloc((size_t)BB * NP * NP * 2); // reused as E after T consumed
  float*    Tbuf   = (float*)   alloc((size_t)NN * NP * 4);
  float*    Ah     = (float*)   alloc((size_t)BB * NN * HH * 4);
  _Float16* AhT    = (_Float16*)alloc((size_t)BB * HH * NP * 2);
  float*    xg     = (float*)   alloc((size_t)BB * NN * KC * HH * 4);
  _Float16* wnode  = (_Float16*)alloc((size_t)NN * KC * HH * HH * 2);
  float*    bnode  = (float*)   alloc((size_t)NN * HH * 4);
  float*    g      = (float*)   alloc((size_t)BN * HH * 4);
  _Float16* g16    = (_Float16*)alloc((size_t)BN * HH * 2);

  auto blks = [](size_t n) { return (unsigned)((n + 255) / 256); };

  // 1) f16 conversions (with K padding where needed)
  k_cvt_pad<<<blks((size_t)BN * CC), 256, 0, stream>>>(z, z16, BN, CC, CC);
  k_cvt_pad<<<blks((size_t)HH * CC), 256, 0, stream>>>(W_in, Win16, HH, CC, CC);
  k_cvt_pad<<<blks((size_t)OC * HH), 256, 0, stream>>>(W_out, Wout16, OC, HH, HH);
  k_cvt_pad<<<blks((size_t)NN * NP), 256, 0, stream>>>(vs, vs16, NN, NN, NP);

  // 2) adjacency A = softmax(relu(emb emb^T)) -> f16
  k_adj<<<NN, 256, 0, stream>>>(emb, A16);

  // 3) h = relu(z @ W_in^T + b_in)   [WMMA, fused bias+relu]
  k_gemm<1><<<dim3(BN / 16, HH / 64, 1), 32, 0, stream>>>(
      z16, CC, 0, Win16, CC, 0, b_in, h, HH, 0, CC);

  // 4) e1/e2 and K-major f16 copy of h
  k_e12<<<blks(BN), 256, 0, stream>>>(h, w1, w2, e1, e2);
  k_colmajor<<<dim3(blks((size_t)HH * NP), BB, 1), 256, 0, stream>>>(h, hT);

  // 5) sigT = sigmoid(e1 ox e2 + bs), K-major, zero padded (NP x NP slab per batch)
  k_sig<<<dim3(blks((size_t)NP * NP), BB, 1), 256, 0, stream>>>(e1, e2, bs, sigT);

  // 6) per batch: T = vs @ sig  (the 128-GFLOP op, WMMA), then row-softmax -> E
  //    E overwrites the (now dead) sigT[b] slab.
  for (int b = 0; b < BB; ++b) {
    _Float16* slab = sigT + (size_t)b * NP * NP;
    k_gemm<0><<<dim3(NN / 16, NP / 64, 1), 32, 0, stream>>>(
        vs16, NP, 0, slab, NP, 0, nullptr, Tbuf, NP, 0, NP);
    k_softmax_T<<<NN, 256, 0, stream>>>(Tbuf, slab);
  }

  // 7) Ah = A @ h (batched over b via Bt), then K-major f16
  k_gemm<0><<<dim3(NN / 16, HH / 64, BB), 32, 0, stream>>>(
      A16, NP, 0, hT, NP, (long long)HH * NP, nullptr,
      Ah, HH, (long long)NN * HH, NP);
  k_colmajor<<<dim3(blks((size_t)HH * NP), BB, 1), 256, 0, stream>>>(Ah, AhT);

  // 8) x_g[...,0,:] = E @ h ; x_g[...,1,:] = E @ Ah   (sup@h eliminated algebraically)
  k_gemm<0><<<dim3(NN / 16, HH / 64, BB), 32, 0, stream>>>(
      sigT, NP, (long long)NP * NP, hT, NP, (long long)HH * NP, nullptr,
      xg, KC * HH, (long long)NN * KC * HH, NP);
  k_gemm<0><<<dim3(NN / 16, HH / 64, BB), 32, 0, stream>>>(
      sigT, NP, (long long)NP * NP, AhT, NP, (long long)HH * NP, nullptr,
      xg + HH, KC * HH, (long long)NN * KC * HH, NP);

  // 9) node-adaptive weights / bias, then g
  k_weights<<<blks((size_t)NN * KC * HH * HH), 256, 0, stream>>>(emb, wp, wnode);
  k_biasnode<<<blks((size_t)NN * HH), 256, 0, stream>>>(emb, bp, bnode);
  k_g<<<BN, 64, 0, stream>>>(xg, wnode, bnode, g, g16);

  // 10) out = 0.1*tanh(g @ W_out^T + b_out)  [WMMA, fused epilogue]
  k_gemm<2><<<dim3(BN / 16, OC / 64, 1), 32, 0, stream>>>(
      g16, HH, 0, Wout16, HH, 0, b_out, out, OC, 0, HH);
}
